// QuantCasualSelfAttention_71365176590950
// MI455X (gfx1250) — matrix operand order repaired and checked by reading, not compiled, not run on hardware
//
#include <hip/hip_runtime.h>
#include <hip/hip_bf16.h>
#include <math.h>

typedef __bf16 bf16;
typedef __attribute__((ext_vector_type(16))) __bf16 v16bf;
typedef __attribute__((ext_vector_type(8)))  __bf16 v8bf;
typedef __attribute__((ext_vector_type(4)))  __bf16 v4bf;
typedef __attribute__((ext_vector_type(8)))  float  v8f;
typedef __attribute__((ext_vector_type(4)))  float  v4f;
typedef unsigned int v4u __attribute__((ext_vector_type(4)));
typedef unsigned int v8u __attribute__((ext_vector_type(8)));

union AFrag { v16bf v; v8bf h[2]; };

__device__ __forceinline__ v8f wmma_bf16(v16bf a, v16bf b, v8f c) {
  return __builtin_amdgcn_wmma_f32_16x16x32_bf16(false, a, false, b,
                                                 (short)0, c, false, false);
}

// LDS byte offset of a __shared__ object (generic LDS pointer low 32 bits).
__device__ __forceinline__ unsigned ldsoff(const void* p) {
  return (unsigned)(unsigned long long)p;
}

// ---------------------------------------------------------------------------
// Tensor Data Mover: async 2D bf16 tile load global -> LDS with row padding.
// Tile = d1 rows of d0 elements (2 B), global row stride `stride` elems.
// pad: 4 DWORDs (16 B) appended after every row; interval code 4 = 32 DWORDs
// (64-elem rows -> 72-elem padded rows, 144 B, 16-B aligned).
// D# per CDNA5 ISA 8.3/8.4.  Tracked by TENSORcnt.
// ---------------------------------------------------------------------------
__device__ __forceinline__ void tdm_load_2d_pad(unsigned lds_byte_off,
                                                const bf16* g, unsigned d0,
                                                unsigned d1,
                                                unsigned long long stride) {
  unsigned long long ga = (unsigned long long)(const void*)g;
  v4u g0;
  g0.x = 1u;                                   // count=1, user descriptor
  g0.y = lds_byte_off;                         // lds_addr
  g0.z = (unsigned)ga;                         // global_addr[31:0]
  g0.w = (unsigned)((ga >> 32) & 0x01FFFFFFu)  // global_addr[56:32]
         | 0x80000000u;                        // type=2 ("image")
  v8u g1;
  g1[0] = (1u << 16)                 // data_size = 1 -> 2 bytes
        | (1u << 20)                 // pad_enable
        | (4u << 22)                 // pad_interval: 4 -> 32 DWORDs (one row)
        | (3u << 25);                // pad_amount: 3 -> 4 DWORDs (16 B)
  g1[1] = (d0 & 0xFFFFu) << 16;                            // tensor_dim0 lo
  g1[2] = ((d0 >> 16) & 0xFFFFu) | ((d1 & 0xFFFFu) << 16); // td0 hi | td1 lo
  g1[3] = ((d1 >> 16) & 0xFFFFu) | ((d0 & 0xFFFFu) << 16); // td1 hi | tile_d0
  g1[4] = d1 & 0xFFFFu;                                    // tile_d1 (d2=0)
  g1[5] = (unsigned)stride;                                // dim0_stride lo
  g1[6] = (unsigned)((stride >> 32) & 0xFFFFu);            // stride hi
  g1[7] = 0u;
  asm volatile("tensor_load_to_lds %0, %1" :: "s"(g0), "s"(g1) : "memory");
}

// ---------------------------------------------------------------------------
// Prep kernel A: f32 -> bf16 elementwise.
// ---------------------------------------------------------------------------
__global__ __launch_bounds__(256) void convert_bf16(
    const float* __restrict__ src, bf16* __restrict__ dst) {
  size_t i = (size_t)blockIdx.x * 256 + threadIdx.x;
  v4f f = *(const v4f*)&src[i * 4];
  v4bf h; h.x = (bf16)f.x; h.y = (bf16)f.y; h.z = (bf16)f.z; h.w = (bf16)f.w;
  *(v4bf*)&dst[i * 4] = h;
}

// ---------------------------------------------------------------------------
// Prep kernel B: W[K][N] f32 -> Wt[N][K] bf16 (32x32 LDS tiles, coalesced).
// ---------------------------------------------------------------------------
__global__ __launch_bounds__(256) void convert_transpose(
    const float* __restrict__ W, bf16* __restrict__ Wt, int K, int N) {
  __shared__ float tile[32][33];
  const int tx = threadIdx.x & 31, ty = threadIdx.x >> 5;
  const int n0 = blockIdx.x * 32, k0 = blockIdx.y * 32;
#pragma unroll
  for (int i = 0; i < 4; ++i)
    tile[ty + 8 * i][tx] = W[(size_t)(k0 + ty + 8 * i) * N + n0 + tx];
  __syncthreads();
#pragma unroll
  for (int i = 0; i < 4; ++i)
    Wt[(size_t)(n0 + ty + 8 * i) * K + k0 + tx] = (bf16)tile[tx][ty + 8 * i];
}

// ---------------------------------------------------------------------------
// Kernel 1: QKV GEMM, TDM double-buffered, K-chunk 64 (16 WMMA / barrier).
// A = xb[4096,1024] bf16, B = Wat[3072,1024] bf16 (pre-transposed).
// Block 128x128, 8 waves of 32x64.  72 KB LDS (ping-pong tiles).
// Outputs: Q,K [B,H,T,D] bf16, V transposed [B,H,D,T] bf16.
// ---------------------------------------------------------------------------
__global__ __launch_bounds__(256) void qkv_gemm(
    const bf16* __restrict__ xb, const bf16* __restrict__ Wat,
    const float* __restrict__ bias,
    bf16* __restrict__ Qb, bf16* __restrict__ Kb, bf16* __restrict__ Vt) {
  __shared__ __align__(16) bf16 As[2][128][72];   // 2 x 18 KB
  __shared__ __align__(16) bf16 Bs[2][128][72];   // 2 x 18 KB

  const int t    = threadIdx.x;
  const int n0   = blockIdx.x * 128;
  const int m0   = blockIdx.y * 128;
  const int wave = t >> 5, lane = t & 31;
  const int hb   = lane >> 4, lrow = lane & 15;
  const int wm   = wave & 3,  wn   = wave >> 2;

  v8f acc[2][4];
#pragma unroll
  for (int i = 0; i < 2; ++i)
#pragma unroll
    for (int j = 0; j < 4; ++j)
#pragma unroll
      for (int e = 0; e < 8; ++e) acc[i][j][e] = 0.f;

  // Prologue: fill buffer 0 (k0 = 0).
  if (wave == 0)
    tdm_load_2d_pad(ldsoff(&As[0][0][0]), xb + (size_t)m0 * 1024, 64, 128,
                    1024);
  else if (wave == 1)
    tdm_load_2d_pad(ldsoff(&Bs[0][0][0]), Wat + (size_t)n0 * 1024, 64, 128,
                    1024);
  __builtin_amdgcn_s_wait_tensorcnt(0);
  __syncthreads();

  for (int it = 0; it < 16; ++it) {
    const int cur = it & 1, nxt = cur ^ 1;
    if (it < 15) {                     // overlap next tile's DMA with WMMAs
      const int k0n = (it + 1) * 64;
      if (wave == 0)
        tdm_load_2d_pad(ldsoff(&As[nxt][0][0]),
                        xb + (size_t)m0 * 1024 + k0n, 64, 128, 1024);
      else if (wave == 1)
        tdm_load_2d_pad(ldsoff(&Bs[nxt][0][0]),
                        Wat + (size_t)n0 * 1024 + k0n, 64, 128, 1024);
    }

#pragma unroll
    for (int kt = 0; kt < 2; ++kt) {
      AFrag a[2], b[4];
#pragma unroll
      for (int mt = 0; mt < 2; ++mt) {
        const v8bf* Ar = (const v8bf*)&As[cur][wm * 32 + mt * 16 + lrow][0];
        a[mt].h[0] = Ar[kt * 4 + hb];          // k = kt*32 + 8*half
        a[mt].h[1] = Ar[kt * 4 + 2 + hb];      // k = kt*32 + 16 + 8*half
      }
#pragma unroll
      for (int nt = 0; nt < 4; ++nt) {
        const v8bf* Br = (const v8bf*)&Bs[cur][wn * 64 + nt * 16 + lrow][0];
        b[nt].h[0] = Br[kt * 4 + hb];          // k = kt*32 + 8*half
        b[nt].h[1] = Br[kt * 4 + 2 + hb];
      }
#pragma unroll
      for (int mt = 0; mt < 2; ++mt)
#pragma unroll
        for (int nt = 0; nt < 4; ++nt)
          acc[mt][nt] = wmma_bf16(a[mt].v, b[nt].v, acc[mt][nt]);
    }

    __builtin_amdgcn_s_wait_tensorcnt(0);   // next buffer landed
    __syncthreads();                        // publish; fences reads of cur
  }

  // Epilogue: bias + scatter.  Section uniform per block (128 | 1024).
  const int sec = n0 >> 10;                  // 0=Q 1=K 2=V
#pragma unroll
  for (int mt = 0; mt < 2; ++mt) {
#pragma unroll
    for (int nt = 0; nt < 4; ++nt) {
      int n = n0 + wn * 64 + nt * 16 + lrow;
      float bv = bias[n];
      int c = n & 1023;
      int hh = c >> 6, d = c & 63;
#pragma unroll
      for (int s = 0; s < 8; ++s) {
        int m = m0 + wm * 32 + mt * 16 + s + hb * 8;
        int bb = m >> 11, tt = m & 2047;
        bf16 val = (bf16)(acc[mt][nt][s] + bv);
        if (sec == 0)
          Qb[(((size_t)bb * 16 + hh) * 2048 + tt) * 64 + d] = val;
        else if (sec == 1)
          Kb[(((size_t)bb * 16 + hh) * 2048 + tt) * 64 + d] = val;
        else
          Vt[(((size_t)bb * 16 + hh) * 64 + d) * 2048 + tt] = val;
      }
    }
  }
}

// ---------------------------------------------------------------------------
// Kernel 2: causal flash attention, TDM double-buffered K/V (72 KB LDS).
// Block = (b, h, 128 q rows); 8 waves x 16 q rows; 64-wide key blocks;
// online softmax; next key block's DMA overlaps current block's compute.
// ---------------------------------------------------------------------------
__global__ __launch_bounds__(256) void attn(
    const bf16* __restrict__ Qb, const bf16* __restrict__ Kb,
    const bf16* __restrict__ Vt, bf16* __restrict__ Y) {
  __shared__ __align__(16) bf16 Qs[128][72];        // 18 KB
  __shared__ __align__(16) bf16 Ks[2][64][72];      // [s][d], 2 x 9 KB
  __shared__ __align__(16) bf16 Vs[2][64][72];      // [d][s], 2 x 9 KB
  __shared__ __align__(16) bf16 Ps[8][16][72];      // per-wave P, 18 KB

  const int t    = threadIdx.x;
  const int qt   = blockIdx.x;
  const int hcur = blockIdx.y;
  const int b    = blockIdx.z;
  const int bh   = b * 16 + hcur;
  const int wave = t >> 5, lane = t & 31, hb = lane >> 4, lrow = lane & 15;
  const size_t qk_base = (size_t)bh * 2048 * 64;

  // Prologue: Q tile + key-block 0 into buffer 0.
  if (wave == 2)       // Q: 128 rows x 64, contiguous rows (stride 64)
    tdm_load_2d_pad(ldsoff(&Qs[0][0]), Qb + qk_base + (size_t)qt * 128 * 64,
                    64, 128, 64);
  else if (wave == 0)  // K tile [s][d], stride 64
    tdm_load_2d_pad(ldsoff(&Ks[0][0][0]), Kb + qk_base, 64, 64, 64);
  else if (wave == 1)  // V tile [d][s], stride 2048
    tdm_load_2d_pad(ldsoff(&Vs[0][0][0]), Vt + qk_base, 64, 64, 2048);
  __builtin_amdgcn_s_wait_tensorcnt(0);
  __syncthreads();

  v8f O[4];
  float m_i[8], l_i[8];
#pragma unroll
  for (int i = 0; i < 4; ++i)
#pragma unroll
    for (int e = 0; e < 8; ++e) O[i][e] = 0.f;
#pragma unroll
  for (int s = 0; s < 8; ++s) { m_i[s] = -INFINITY; l_i[s] = 0.f; }

  const int qrow0 = qt * 128 + wave * 16;
  const int nblk  = (qt * 128 + 128) / 64;     // causal: stop after q tile
  const float scale = 0.125f;                  // 1/sqrt(64)

  for (int ib = 0; ib < nblk; ++ib) {
    const int cur = ib & 1, nxt = cur ^ 1;
    const int s0 = ib * 64;
    if (ib + 1 < nblk) {               // overlap next key block's DMA
      const int s0n = s0 + 64;
      if (wave == 0)
        tdm_load_2d_pad(ldsoff(&Ks[nxt][0][0]),
                        Kb + qk_base + (size_t)s0n * 64, 64, 64, 64);
      else if (wave == 1)
        tdm_load_2d_pad(ldsoff(&Vs[nxt][0][0]),
                        Vt + qk_base + s0n, 64, 64, 2048);
    }

    // S = Q K^T  (wave: 16 x 64)
    v8f S[4];
#pragma unroll
    for (int nt = 0; nt < 4; ++nt)
#pragma unroll
      for (int e = 0; e < 8; ++e) S[nt][e] = 0.f;
#pragma unroll
    for (int kt = 0; kt < 2; ++kt) {
      AFrag a;
      const v8bf* Ar = (const v8bf*)&Qs[wave * 16 + lrow][0];
      a.h[0] = Ar[kt * 4 + hb];
      a.h[1] = Ar[kt * 4 + 2 + hb];
#pragma unroll
      for (int nt = 0; nt < 4; ++nt) {
        AFrag bfr;
        const v8bf* Br = (const v8bf*)&Ks[cur][nt * 16 + lrow][0];
        bfr.h[0] = Br[kt * 4 + hb];
        bfr.h[1] = Br[kt * 4 + 2 + hb];
        S[nt] = wmma_bf16(a.v, bfr.v, S[nt]);
      }
    }

    // causal mask + scale + online softmax (row = slot + 8*half)
    float mnew[8];
#pragma unroll
    for (int s = 0; s < 8; ++s) {
      int qg = qrow0 + s + hb * 8;
      float mx = -INFINITY;
#pragma unroll
      for (int nt = 0; nt < 4; ++nt) {
        int sg = s0 + nt * 16 + lrow;
        float v = S[nt][s] * scale;
        v = (sg <= qg) ? v : -INFINITY;
        S[nt][s] = v;
        mx = fmaxf(mx, v);
      }
      mx = fmaxf(mx, __shfl_xor(mx, 1, 32));
      mx = fmaxf(mx, __shfl_xor(mx, 2, 32));
      mx = fmaxf(mx, __shfl_xor(mx, 4, 32));
      mx = fmaxf(mx, __shfl_xor(mx, 8, 32));   // stays within 16-lane halves
      mnew[s] = fmaxf(m_i[s], mx);
    }
#pragma unroll
    for (int s = 0; s < 8; ++s) {
      float rs = 0.f;
#pragma unroll
      for (int nt = 0; nt < 4; ++nt) {
        float p = __expf(S[nt][s] - mnew[s]);
        rs += p;
        Ps[wave][s + hb * 8][nt * 16 + lrow] = (bf16)p;
      }
      rs += __shfl_xor(rs, 1, 32);
      rs += __shfl_xor(rs, 2, 32);
      rs += __shfl_xor(rs, 4, 32);
      rs += __shfl_xor(rs, 8, 32);
      float alpha = __expf(m_i[s] - mnew[s]);
      l_i[s] = l_i[s] * alpha + rs;
      m_i[s] = mnew[s];
#pragma unroll
      for (int dt = 0; dt < 4; ++dt) O[dt][s] *= alpha;
    }

    // O += P * V  (wave-private LDS round-trip re-fragments P into A layout)
#pragma unroll
    for (int kt = 0; kt < 2; ++kt) {
      AFrag a;
      const v8bf* Ar = (const v8bf*)&Ps[wave][lrow][0];
      a.h[0] = Ar[kt * 4 + hb];
      a.h[1] = Ar[kt * 4 + 2 + hb];
#pragma unroll
      for (int dt = 0; dt < 4; ++dt) {
        AFrag bfr;
        const v8bf* Br = (const v8bf*)&Vs[cur][dt * 16 + lrow][0];
        bfr.h[0] = Br[kt * 4 + hb];
        bfr.h[1] = Br[kt * 4 + 2 + hb];
        O[dt] = wmma_bf16(a.v, bfr.v, O[dt]);
      }
    }

    __builtin_amdgcn_s_wait_tensorcnt(0);   // next K/V buffer landed
    __syncthreads();                        // publish; fences reads of cur
  }

  // normalize + store Y bf16 [B,T,C]
#pragma unroll
  for (int s = 0; s < 8; ++s) {
    float inv = 1.f / l_i[s];
    int qg = qrow0 + s + hb * 8;
    size_t rowbase = ((size_t)b * 2048 + qg) * 1024 + hcur * 64;
#pragma unroll
    for (int dt = 0; dt < 4; ++dt)
      Y[rowbase + dt * 16 + lrow] = (bf16)(O[dt][s] * inv);
  }
}

// ---------------------------------------------------------------------------
// Kernel 3: output projection, TDM double-buffered, K-chunk 64.
// Yb[4096,1024] bf16 * Wpt^T + bias -> f32 out.
// ---------------------------------------------------------------------------
__global__ __launch_bounds__(256) void proj_gemm(
    const bf16* __restrict__ Yb, const bf16* __restrict__ Wpt,
    const float* __restrict__ bias, float* __restrict__ out) {
  __shared__ __align__(16) bf16 As[2][128][72];
  __shared__ __align__(16) bf16 Bs[2][128][72];

  const int t    = threadIdx.x;
  const int n0   = blockIdx.x * 128;
  const int m0   = blockIdx.y * 128;
  const int wave = t >> 5, lane = t & 31;
  const int hb   = lane >> 4, lrow = lane & 15;
  const int wm   = wave & 3,  wn   = wave >> 2;

  v8f acc[2][4];
#pragma unroll
  for (int i = 0; i < 2; ++i)
#pragma unroll
    for (int j = 0; j < 4; ++j)
#pragma unroll
      for (int e = 0; e < 8; ++e) acc[i][j][e] = 0.f;

  if (wave == 0)
    tdm_load_2d_pad(ldsoff(&As[0][0][0]), Yb + (size_t)m0 * 1024, 64, 128,
                    1024);
  else if (wave == 1)
    tdm_load_2d_pad(ldsoff(&Bs[0][0][0]), Wpt + (size_t)n0 * 1024, 64, 128,
                    1024);
  __builtin_amdgcn_s_wait_tensorcnt(0);
  __syncthreads();

  for (int it = 0; it < 16; ++it) {
    const int cur = it & 1, nxt = cur ^ 1;
    if (it < 15) {
      const int k0n = (it + 1) * 64;
      if (wave == 0)
        tdm_load_2d_pad(ldsoff(&As[nxt][0][0]),
                        Yb + (size_t)m0 * 1024 + k0n, 64, 128, 1024);
      else if (wave == 1)
        tdm_load_2d_pad(ldsoff(&Bs[nxt][0][0]),
                        Wpt + (size_t)n0 * 1024 + k0n, 64, 128, 1024);
    }

#pragma unroll
    for (int kt = 0; kt < 2; ++kt) {
      AFrag a[2], b[4];
#pragma unroll
      for (int mt = 0; mt < 2; ++mt) {
        const v8bf* Ar = (const v8bf*)&As[cur][wm * 32 + mt * 16 + lrow][0];
        a[mt].h[0] = Ar[kt * 4 + hb];
        a[mt].h[1] = Ar[kt * 4 + 2 + hb];
      }
#pragma unroll
      for (int nt = 0; nt < 4; ++nt) {
        const v8bf* Br = (const v8bf*)&Bs[cur][wn * 64 + nt * 16 + lrow][0];
        b[nt].h[0] = Br[kt * 4 + hb];
        b[nt].h[1] = Br[kt * 4 + 2 + hb];
      }
#pragma unroll
      for (int mt = 0; mt < 2; ++mt)
#pragma unroll
        for (int nt = 0; nt < 4; ++nt)
          acc[mt][nt] = wmma_bf16(a[mt].v, b[nt].v, acc[mt][nt]);
    }

    __builtin_amdgcn_s_wait_tensorcnt(0);
    __syncthreads();
  }

#pragma unroll
  for (int mt = 0; mt < 2; ++mt) {
#pragma unroll
    for (int nt = 0; nt < 4; ++nt) {
      int n = n0 + wn * 64 + nt * 16 + lrow;
      float bv = bias[n];
#pragma unroll
      for (int s = 0; s < 8; ++s) {
        int m = m0 + wm * 32 + mt * 16 + s + hb * 8;
        out[(size_t)m * 1024 + n] = acc[mt][nt][s] + bv;
      }
    }
  }
}

// ---------------------------------------------------------------------------
extern "C" void kernel_launch(void* const* d_in, const int* in_sizes, int n_in,
                              void* d_out, int out_size, void* d_ws,
                              size_t ws_size, hipStream_t stream) {
  const float* x      = (const float*)d_in[0];
  const float* W_attn = (const float*)d_in[1];
  const float* b_attn = (const float*)d_in[2];
  const float* W_proj = (const float*)d_in[3];
  const float* b_proj = (const float*)d_in[4];
  float* out = (float*)d_out;

  // Workspace (bf16 elems): xb 4M | Wat 3M | Wpt 1M | Qb 4M | Kb 4M | Vt 4M
  // | Yb 4M  = 24M (48 MB)
  bf16* xb  = (bf16*)d_ws;
  bf16* Wat = xb  + (size_t)4 * 1024 * 1024;
  bf16* Wpt = Wat + (size_t)3 * 1024 * 1024;
  bf16* Qb  = Wpt + (size_t)1 * 1024 * 1024;
  bf16* Kb  = Qb  + (size_t)4 * 1024 * 1024;
  bf16* Vt  = Kb  + (size_t)4 * 1024 * 1024;
  bf16* Yb  = Vt  + (size_t)4 * 1024 * 1024;

  convert_bf16<<<4096, 256, 0, stream>>>(x, xb);
  dim3 gt1(3072 / 32, 1024 / 32);
  convert_transpose<<<gt1, 256, 0, stream>>>(W_attn, Wat, 1024, 3072);
  dim3 gt2(1024 / 32, 1024 / 32);
  convert_transpose<<<gt2, 256, 0, stream>>>(W_proj, Wpt, 1024, 1024);

  dim3 g1(3072 / 128, 4096 / 128);
  qkv_gemm<<<g1, 256, 0, stream>>>(xb, Wat, b_attn, Qb, Kb, Vt);

  dim3 g2(2048 / 128, 16, 2);
  attn<<<g2, 256, 0, stream>>>(Qb, Kb, Vt, Yb);

  dim3 g3(1024 / 128, 4096 / 128);
  proj_gemm<<<g3, 256, 0, stream>>>(Yb, Wpt, b_proj, out);
}
